// DTLN_NS_P2_53429393162678
// MI455X (gfx1250) — matrix operand order repaired and checked by reading, not compiled, not run on hardware
//
#include <hip/hip_runtime.h>
#include <hip/hip_bf16.h>
#include <math.h>

// ---------------------------------------------------------------------------
// DTLN noise suppressor for MI455X (gfx1250), wave32 + WMMA f32_16x16x32_f16.
// Shapes: B=64, T=1024, FRAME=512, ENC=256, HID=512, 4H=2048.
// ---------------------------------------------------------------------------

#define B_   64
#define T_   1024
#define FRAME_ 512
#define ENC_ 256
#define HID_ 512
#define G4H_ 2048

typedef __attribute__((ext_vector_type(16))) _Float16 v16h;
typedef __attribute__((ext_vector_type(8)))  _Float16 v8h;
typedef __attribute__((ext_vector_type(8)))  float    v8f;

// WMMA fragment layouts (CDNA5 ISA 7.12.2, wave32):
//  A 16x32 f16 : lane l -> m=l&15 ; elem e -> k = (l>>4)*16 + e (this code's order; ISA table differs)
//  B 32x16 f16 : lane l -> n=l&15 ; elem e -> k = (l>>4)*16 + e (this code's order; ISA table differs)
//  C/D 16x16 f32: lane l -> n=l&15 ; vgpr r -> m = r + 8*(l>>4)

__device__ __forceinline__ v8f wmma_f16(v16h a, v16h b, v8f c) {
    return __builtin_amdgcn_wmma_f32_16x16x32_f16(false, a, false, b,
                                                  (short)0, c, false, false);
}

// A fragment from row-major f16 [M,K] in global memory.
__device__ __forceinline__ v16h load_a_frag(const _Float16* A, int K,
                                            int row0, int k0, int lane) {
    const _Float16* p0 = A + (size_t)(row0 + (lane & 15)) * K + k0 + ((lane >> 4) * 16);
    return *(const v16h*)p0;
}

// A fragment from a plain f16 [16][ldk] tile in LDS.
__device__ __forceinline__ v16h load_a_frag_lds(const _Float16* tile, int ldk,
                                                int k0, int lane) {
    const _Float16* p0 = tile + (lane & 15) * ldk + k0 + ((lane >> 4) * 16);
    v16h r;
#pragma unroll
    for (int i = 0; i < 16; ++i) { r[i] = p0[i]; }
    return r;
}

// B fragment from pre-swizzled weights: one contiguous 32B load per lane.
__device__ __forceinline__ v16h load_b_frag(const _Float16* Wswz, int blk, int lane) {
    return *(const v16h*)(Wswz + ((size_t)blk * 32 + lane) * 16);
}

__device__ __forceinline__ float sigf(float x) { return 1.f / (1.f + __expf(-x)); }

// ---------------------------------------------------------------------------
// Prep: f32 weight [N,K] (row-major, W[n][k]) -> f16 in B-fragment order:
//   dst[((ntile*(K/32)+ks)*32 + lane)*16 + e] = W[ntile*16+(lane&15)][ks*32+(lane>>4)*16+e]
// ---------------------------------------------------------------------------
__global__ void swizzle_weight_kernel(const float* __restrict__ src,
                                      _Float16* __restrict__ dst, int N, int K) {
    int idx = blockIdx.x * blockDim.x + threadIdx.x;   // one per (blk,lane)
    int ksteps = K >> 5;
    int total = (N >> 4) * ksteps * 32;
    if (idx >= total) return;
    int lane = idx & 31;
    int blk  = idx >> 5;
    int ks   = blk % ksteps;
    int nt   = blk / ksteps;
    const float* s = src + (size_t)(nt * 16 + (lane & 15)) * K + ks * 32 + (lane >> 4) * 16;
    _Float16* d = dst + (size_t)idx * 16;
#pragma unroll
    for (int i = 0; i < 16; ++i) d[i] = (_Float16)s[i];
}

__global__ void bias_sum_kernel(const float* a, const float* b, float* o, int n) {
    int i = blockIdx.x * blockDim.x + threadIdx.x;
    if (i < n) o[i] = a[i] + b[i];
}

// ---------------------------------------------------------------------------
// Encoder GEMM + fused LayerNorm.  One WG = 16 t-rows x 256 ENC cols (a full
// LN row).  16 waves, wave w owns o-tile w.  K = FRAME = 512 -> 16 k-steps,
// A tile double-buffered in LDS (one barrier per k-step).
// ---------------------------------------------------------------------------
__global__ __launch_bounds__(512) void enc_ln_kernel(
        const float* __restrict__ x, const _Float16* __restrict__ WencSwz,
        const float* __restrict__ ln_g, const float* __restrict__ ln_b,
        _Float16* __restrict__ enc_raw, _Float16* __restrict__ enc_nrm) {
    __shared__ _Float16 aT[2][16][32];     // double-buffered A tile [t][c]
    __shared__ float    cT[16][ENC_ + 8];  // full C tile for LN

    int b  = blockIdx.x >> 6;              // T/16 = 64 tiles per batch
    int t0 = (blockIdx.x & 63) << 4;
    int tid = threadIdx.x, lane = tid & 31, wave = tid >> 5;
    int c_l = tid >> 4, t_l = tid & 15;

    const size_t xbase = (size_t)b * FRAME_ * T_ + t0 + t_l;
    // stage k-step 0
    aT[0][t_l][c_l] = (_Float16)x[xbase + (size_t)c_l * T_];

    v8f acc = {};
#pragma unroll 1
    for (int ks = 0; ks < 16; ++ks) {
        __syncthreads();
        v16h a = load_a_frag_lds(&aT[ks & 1][0][0], 32, 0, lane);
        if (ks + 1 < 16)   // stage next tile into the other buffer
            aT[(ks + 1) & 1][t_l][c_l] =
                (_Float16)x[xbase + (size_t)((ks + 1) * 32 + c_l) * T_];
        v16h bf = load_b_frag(WencSwz, wave * 16 + ks, lane);
        acc = wmma_f16(a, bf, acc);
    }
    // spill C frag to LDS for the row-wise LayerNorm
    {
        int n = wave * 16 + (lane & 15);
        int rb = (lane >> 4) * 8;
#pragma unroll
        for (int r = 0; r < 8; ++r) cT[rb + r][n] = acc[r];
    }
    __syncthreads();
    // wave w normalizes row w (256 values, 8 per lane, wave32 reduce)
    {
        int row = wave;
        float s = 0.f, ss = 0.f;
#pragma unroll
        for (int j = 0; j < 8; ++j) { float v = cT[row][lane + 32 * j]; s += v; ss += v * v; }
#pragma unroll
        for (int off = 16; off >= 1; off >>= 1) {
            s  += __shfl_xor(s,  off, 32);
            ss += __shfl_xor(ss, off, 32);
        }
        float mu   = s * (1.f / ENC_);
        float var  = ss * (1.f / ENC_) - mu * mu;
        float rstd = rsqrtf(var + 1e-5f);
        size_t obase = ((size_t)b * T_ + t0 + row) * ENC_;
#pragma unroll
        for (int j = 0; j < 8; ++j) {
            int o = lane + 32 * j;
            float v = cT[row][o];
            enc_raw[obase + o] = (_Float16)v;
            enc_nrm[obase + o] = (_Float16)((v - mu) * rstd * ln_g[o] + ln_b[o]);
        }
    }
}

// ---------------------------------------------------------------------------
// Generic WMMA GEMM: C[M,N] = A[M,K](f16,row-major) * W^T + bias, W swizzled.
// WG tile 32m x 256n, 16 waves: wave w -> (mt=w&1, 2 n-tiles at (w>>1)*32).
// EPI 0: store f16 (acc+bias) at [T,B,N] (time-major, LSTM pre-activations)
// EPI 1: store f16 sigmoid(acc+bias)*aux[row,col]  (masked encoding, [M,N])
// EPI 2: store f32, transposed into decoded[B,FRAME,T] (vectorized over t)
// ---------------------------------------------------------------------------
template <int EPI>
__global__ __launch_bounds__(512) void gemm_wmma_kernel(
        const _Float16* __restrict__ A, const _Float16* __restrict__ Wswz,
        const float* __restrict__ bias, void* __restrict__ out,
        const _Float16* __restrict__ aux, int M, int N, int K, int Tdim) {
    int tid = threadIdx.x, lane = tid & 31, wave = tid >> 5;
    int nBlocks = N >> 8;
    int bm = blockIdx.x / nBlocks, bn = blockIdx.x % nBlocks;
    int row0 = bm * 32 + (wave & 1) * 16;
    int col0 = bn * 256 + (wave >> 1) * 32;
    int ksteps = K >> 5;
    int ntile = col0 >> 4;

    v8f acc[2] = {{}, {}};
#pragma unroll 1
    for (int ks = 0; ks < ksteps; ++ks) {
        v16h a = load_a_frag(A, K, row0, ks * 32, lane);
#pragma unroll
        for (int j = 0; j < 2; ++j) {
            v16h bf = load_b_frag(Wswz, (ntile + j) * ksteps + ks, lane);
            acc[j] = wmma_f16(a, bf, acc[j]);
        }
    }
    int rb = (lane >> 4) * 8;
#pragma unroll
    for (int j = 0; j < 2; ++j) {
        int col = col0 + j * 16 + (lane & 15);
        if (EPI == 2) {
            // 8 accumulator rows == 8 consecutive t -> two float4 stores
            int rowb = row0 + rb;
            int t0r = rowb % Tdim, bb = rowb / Tdim;
            float* p = (float*)out + ((size_t)bb * N + col) * Tdim + t0r;
            *(float4*)p       = make_float4(acc[j][0], acc[j][1], acc[j][2], acc[j][3]);
            *(float4*)(p + 4) = make_float4(acc[j][4], acc[j][5], acc[j][6], acc[j][7]);
        } else {
            float bv = bias ? bias[col] : 0.f;
#pragma unroll
            for (int r = 0; r < 8; ++r) {
                int row = row0 + rb + r;
                float v = acc[j][r] + bv;
                if (EPI == 0) {   // time-major [T, B, N] for the LSTM
                    int t = row % Tdim, bb = row / Tdim;
                    ((_Float16*)out)[((size_t)t * (M / Tdim) + bb) * N + col] = (_Float16)v;
                } else {          // EPI == 1
                    size_t idx = (size_t)row * N + col;
                    ((_Float16*)out)[idx] = (_Float16)(sigf(v) * (float)aux[idx]);
                }
            }
        }
    }
}

// ---------------------------------------------------------------------------
// Persistent LSTM layer. 4 WGs, each owns 16 batch rows; h(f16) and c(f32)
// live in LDS for all 1024 steps.  Per step: raw = h @ Whh^T via WMMA
// (16 waves x 8 n-tiles x 16 k-steps = 128 WMMA / wave / step), then the
// i,f,g,o nonlinearity adds the time-major precomputed input projection with
// fully coalesced loads.  Whh' (2 MB f16) streams from the 192 MB L2.
// ---------------------------------------------------------------------------
__global__ __launch_bounds__(512) void lstm_kernel(
        const _Float16* __restrict__ pre,      // [T, B, 4H] time-major
        const _Float16* __restrict__ WhhSwz,
        const float* __restrict__ states_in, float* __restrict__ states_out,
        _Float16* __restrict__ xout, int layer) {
    extern __shared__ char smem[];
    _Float16* hS = (_Float16*)smem;                                // [16][512]  16 KB
    float*    cS = (float*)(smem + 16 * HID_ * 2);                 // [16][512]  32 KB
    float*    gS = (float*)(smem + 16 * HID_ * 2 + 16 * HID_ * 4); // [16][2048] 128 KB

    int tid = threadIdx.x, lane = tid & 31, wave = tid >> 5;
    int b0 = blockIdx.x * 16;

    const float* h_in = states_in + (size_t)(2 * layer) * B_ * HID_;
    const float* c_in = h_in + (size_t)B_ * HID_;
    for (int e = tid; e < 16 * HID_; e += 512) {
        int br = e >> 9, hc = e & 511;
        hS[e] = (_Float16)h_in[(size_t)(b0 + br) * HID_ + hc];
        cS[e] = c_in[(size_t)(b0 + br) * HID_ + hc];
    }
    __syncthreads();

    int colbase = wave * 128;      // wave w owns gate cols [128w, 128w+128)
    int ebr = tid >> 5, ehc_0 = (tid & 31);   // elementwise: row=wave, cols lane+32*j
#pragma unroll 1
    for (int t = 0; t < T_; ++t) {
        // prefetch this step's input-projection slab while the GEMM runs
        const _Float16* preRow = pre + ((size_t)t * B_ + b0) * G4H_;
        __builtin_prefetch(preRow + ebr * G4H_ + ehc_0 * 8, 0, 1);

        v8f acc[8] = {{}, {}, {}, {}, {}, {}, {}, {}};
        // raw gates = h @ Whh^T
#pragma unroll 1
        for (int ks = 0; ks < 16; ++ks) {
            v16h a = load_a_frag_lds(hS, HID_, ks * 32, lane);
#pragma unroll
            for (int j = 0; j < 8; ++j) {
                v16h bf = load_b_frag(WhhSwz, ((colbase >> 4) + j) * 16 + ks, lane);
                acc[j] = wmma_f16(a, bf, acc[j]);
            }
        }
        // raw gates to LDS for cross-gate recombination
#pragma unroll
        for (int j = 0; j < 8; ++j) {
            int col = colbase + j * 16 + (lane & 15);
            int rb  = (lane >> 4) * 8;
#pragma unroll
            for (int r = 0; r < 8; ++r) gS[(rb + r) * G4H_ + col] = acc[j][r];
        }
        __syncthreads();
        // elementwise LSTM cell (PyTorch gate order i,f,g,o); pre-activation
        // input projection added here with coalesced time-major loads.
        for (int e = tid; e < 16 * HID_; e += 512) {
            int br = e >> 9, hc = e & 511;
            const float* g = gS + br * G4H_;
            const _Float16* p = preRow + (size_t)br * G4H_;
            float ig = g[hc]         + (float)p[hc];
            float fg = g[512 + hc]   + (float)p[512 + hc];
            float gg = g[1024 + hc]  + (float)p[1024 + hc];
            float og = g[1536 + hc]  + (float)p[1536 + hc];
            float c = sigf(fg) * cS[e] + sigf(ig) * tanhf(gg);
            float h = sigf(og) * tanhf(c);
            cS[e] = c;
            hS[e] = (_Float16)h;
            xout[((size_t)(b0 + br) * T_ + t) * HID_ + hc] = (_Float16)h;
        }
        __syncthreads();
    }
    // final states -> d_out tail ([4,B,HID]: h at 2*layer, c at 2*layer+1)
    float* h_out = states_out + (size_t)(2 * layer) * B_ * HID_;
    float* c_out = h_out + (size_t)B_ * HID_;
    for (int e = tid; e < 16 * HID_; e += 512) {
        int br = e >> 9, hc = e & 511;
        h_out[(size_t)(b0 + br) * HID_ + hc] = (float)hS[e];
        c_out[(size_t)(b0 + br) * HID_ + hc] = cS[e];
    }
}

// ---------------------------------------------------------------------------
// Host-side orchestration
// ---------------------------------------------------------------------------
static inline size_t align256(size_t x) { return (x + 255) & ~(size_t)255; }

extern "C" void kernel_launch(void* const* d_in, const int* in_sizes, int n_in,
                              void* d_out, int out_size, void* d_ws, size_t ws_size,
                              hipStream_t stream) {
    (void)in_sizes; (void)n_in; (void)out_size; (void)ws_size;

    const float* x      = (const float*)d_in[0];
    const float* states = (const float*)d_in[1];
    const float* W_enc  = (const float*)d_in[2];
    const float* ln_g   = (const float*)d_in[3];
    const float* ln_b   = (const float*)d_in[4];
    const float* Wih1   = (const float*)d_in[5];
    const float* Whh1   = (const float*)d_in[6];
    const float* bih1   = (const float*)d_in[7];
    const float* bhh1   = (const float*)d_in[8];
    const float* Wih2   = (const float*)d_in[9];
    const float* Whh2   = (const float*)d_in[10];
    const float* bih2   = (const float*)d_in[11];
    const float* bhh2   = (const float*)d_in[12];
    const float* Wlin   = (const float*)d_in[13];
    const float* blin   = (const float*)d_in[14];
    const float* W_dec  = (const float*)d_in[15];

    float* decoded    = (float*)d_out;                       // [B,FRAME,T]
    float* states_out = decoded + (size_t)B_ * FRAME_ * T_;  // [4,B,HID]

    // ---- workspace carve-up ----
    char* ws = (char*)d_ws;
    size_t off = 0;
    auto carve = [&](size_t bytes) { char* p = ws + off; off += align256(bytes); return p; };
    _Float16* WencS = (_Float16*)carve((size_t)ENC_ * FRAME_ * 2);
    _Float16* Wih1S = (_Float16*)carve((size_t)G4H_ * ENC_ * 2);
    _Float16* Whh1S = (_Float16*)carve((size_t)G4H_ * HID_ * 2);
    _Float16* Wih2S = (_Float16*)carve((size_t)G4H_ * HID_ * 2);
    _Float16* Whh2S = (_Float16*)carve((size_t)G4H_ * HID_ * 2);
    _Float16* WlinS = (_Float16*)carve((size_t)ENC_ * HID_ * 2);
    _Float16* WdecS = (_Float16*)carve((size_t)FRAME_ * ENC_ * 2);
    float*    b1sum = (float*)carve((size_t)G4H_ * 4);
    float*    b2sum = (float*)carve((size_t)G4H_ * 4);
    _Float16* encRaw = (_Float16*)carve((size_t)B_ * T_ * ENC_ * 2);
    _Float16* encNrm = (_Float16*)carve((size_t)B_ * T_ * ENC_ * 2);  // later reused as est
    _Float16* pre    = (_Float16*)carve((size_t)B_ * T_ * G4H_ * 2);  // shared by layer 1/2
    _Float16* x1     = (_Float16*)carve((size_t)B_ * T_ * HID_ * 2);
    _Float16* x2     = (_Float16*)carve((size_t)B_ * T_ * HID_ * 2);

    // ---- weight prep (f32 -> f16, WMMA B-fragment order) ----
    auto swz = [&](const float* src, _Float16* dst, int N, int K) {
        int total = (N >> 4) * (K >> 5) * 32;
        swizzle_weight_kernel<<<(total + 255) / 256, 256, 0, stream>>>(src, dst, N, K);
    };
    swz(W_enc, WencS, ENC_, FRAME_);
    swz(Wih1,  Wih1S, G4H_, ENC_);
    swz(Whh1,  Whh1S, G4H_, HID_);
    swz(Wih2,  Wih2S, G4H_, HID_);
    swz(Whh2,  Whh2S, G4H_, HID_);
    swz(Wlin,  WlinS, ENC_, HID_);
    swz(W_dec, WdecS, FRAME_, ENC_);
    bias_sum_kernel<<<(G4H_ + 255) / 256, 256, 0, stream>>>(bih1, bhh1, b1sum, G4H_);
    bias_sum_kernel<<<(G4H_ + 255) / 256, 256, 0, stream>>>(bih2, bhh2, b2sum, G4H_);

    const int M = B_ * T_;  // 65536

    // ---- encoder + LayerNorm ----
    enc_ln_kernel<<<B_ * (T_ / 16), 512, 0, stream>>>(x, WencS, ln_g, ln_b, encRaw, encNrm);

    // ---- LSTM layer 1 ----
    gemm_wmma_kernel<0><<<(M / 32) * (G4H_ / 256), 512, 0, stream>>>(
        encNrm, Wih1S, b1sum, pre, nullptr, M, G4H_, ENC_, T_);
    const int lstmLds = 16 * HID_ * 2 + 16 * HID_ * 4 + 16 * G4H_ * 4;  // 176 KB
    hipFuncSetAttribute((const void*)lstm_kernel,
                        hipFuncAttributeMaxDynamicSharedMemorySize, lstmLds);
    lstm_kernel<<<B_ / 16, 512, lstmLds, stream>>>(pre, Whh1S, states, states_out, x1, 0);

    // ---- LSTM layer 2 ----
    gemm_wmma_kernel<0><<<(M / 32) * (G4H_ / 256), 512, 0, stream>>>(
        x1, Wih2S, b2sum, pre, nullptr, M, G4H_, HID_, T_);
    lstm_kernel<<<B_ / 16, 512, lstmLds, stream>>>(pre, Whh2S, states, states_out, x2, 1);

    // ---- mask + apply to un-normalized encoding (est reuses encNrm) ----
    gemm_wmma_kernel<1><<<(M / 32) * (ENC_ / 256), 512, 0, stream>>>(
        x2, WlinS, blin, encNrm, encRaw, M, ENC_, HID_, T_);

    // ---- decoder, transposed f32 store into decoded[B,FRAME,T] ----
    gemm_wmma_kernel<2><<<(M / 32) * (FRAME_ / 256), 512, 0, stream>>>(
        encNrm, WdecS, nullptr, decoded, nullptr, M, FRAME_, ENC_, T_);
}
